// UnifiedTransformerHead_46196668235903
// MI455X (gfx1250) — compile-verified, matrix operand order checked by reading, not run on hardware
//
#include <hip/hip_runtime.h>
#include <hip/hip_bf16.h>

typedef __attribute__((ext_vector_type(16))) _Float16 v16h;
typedef __attribute__((ext_vector_type(8)))  _Float16 v8h;
typedef __attribute__((ext_vector_type(4)))  _Float16 v4h;
typedef __attribute__((ext_vector_type(8)))  float    v8f;

#define B_    8
#define D_    256
#define NH_   8
#define NQ_   515
#define LEN_  6380
#define NSEG_ 256
#define NDET_ 256
#define NPT_  3

// ---------------------------------------------------------------------------
// Generic strided/batched WMMA GEMM:  C = act(alpha * A @ B^T + bias)
//   A element (m,k): A + b*sAb + h*sAh + m*lda_m + k*lda_k   (f32)
//   B element (n,k): B + b*sBb + h*sBh + n*ldb_n + k*ldb_k   (f32)
//   C element (m,n): C + b*sCb + h*sCh + m*ldc   + n         (f32)
// block = 256 threads (8 wave32s), tile 64x64, K stepped by 32 through LDS.
// Each wave owns a 32x16 output patch = 2 WMMA tiles (2 v_wmma per K step).
// Staging policy: out-of-range M/N rows are CLAMPED (their C rows/cols are
// never stored, so no zeroing needed -> loads stay branch-free & batched);
// only the K tail needs zeros, done with an unconditional data store followed
// by a conditional zero store (load result always consumed -> no predication).
// Full-K steps stage with float4 (b128) global loads + b64 LDS stores; both
// A and B live in LDS as [row][k] so fragments are 2x ds_load_b128 each.
// ---------------------------------------------------------------------------
__global__ void __launch_bounds__(256) wmma_gemm_kernel(
    const float* __restrict__ A, const float* __restrict__ Bm,
    const float* __restrict__ bias, float* __restrict__ C,
    int M, int N, int K,
    long lda_m, long lda_k, long ldb_n, long ldb_k, long ldc,
    int nh, long sAb, long sAh, long sBb, long sBh, long sCb, long sCh,
    float alpha, int act)
{
  int z = blockIdx.z; int b = z / nh, h = z - b * nh;
  A  += (long)b * sAb + (long)h * sAh;
  Bm += (long)b * sBb + (long)h * sBh;
  C  += (long)b * sCb + (long)h * sCh;

  int m0 = blockIdx.y * 64, n0 = blockIdx.x * 64;
  __shared__ _Float16 sA[64][40];   // [m][k], 80B rows -> 16B-aligned subrows
  __shared__ _Float16 sB[64][40];   // [n][k] (B transposed at stage time)

  int tid  = threadIdx.x;
  int lane = tid & 31, wv = tid >> 5;
  int wc = wv & 3;        // column tile (16 cols each)
  int wr = wv >> 2;       // row half (32 rows each)

  bool vecA  = (lda_k == 1) && ((lda_m & 3) == 0);
  bool vecBk = (ldb_k == 1) && ((ldb_n & 3) == 0);
  bool vecBn = (ldb_n == 1) && ((ldb_k & 3) == 0) && ((N & 3) == 0);

  v8f acc0, acc1;
  #pragma unroll
  for (int i = 0; i < 8; ++i) { acc0[i] = 0.f; acc1[i] = 0.f; }

  for (int k0 = 0; k0 < K; k0 += 32) {
    bool fullK = (k0 + 32 <= K);
    __syncthreads();
    // ---- stage A (64 rows x 32 k) ----
    if (vecA && fullK) {
      #pragma unroll
      for (int j = 0; j < 2; ++j) {
        int f = tid * 2 + j;                  // 0..511 float4 groups
        int row = f >> 3, c4 = (f & 7) * 4;
        int gm = m0 + row; gm = gm < M ? gm : M - 1;   // clamp, no zero
        float4 v = *(const float4*)(A + (long)gm * lda_m + (k0 + c4));
        v4h hv; hv[0] = (_Float16)v.x; hv[1] = (_Float16)v.y;
        hv[2] = (_Float16)v.z; hv[3] = (_Float16)v.w;
        *(v4h*)&sA[row][c4] = hv;
      }
    } else {
      #pragma unroll
      for (int r = 0; r < 8; ++r) {
        int i = r * 256 + tid;
        int row = i >> 5, col = i & 31;
        int gm = m0 + row; gm = gm < M ? gm : M - 1;
        int gk = k0 + col;
        int ck = gk < K ? gk : K - 1;
        float av = A[(long)gm * lda_m + (long)ck * lda_k];
        sA[row][col] = (_Float16)av;          // unconditional: load not sinkable
        if (gk >= K) sA[row][col] = (_Float16)0.f;  // cheap conditional store
      }
    }
    // ---- stage B transposed (64 n-rows x 32 k) ----
    if (vecBk && fullK) {
      #pragma unroll
      for (int j = 0; j < 2; ++j) {
        int f = tid * 2 + j;
        int row = f >> 3, c4 = (f & 7) * 4;   // row = n, c4 = k
        int gn = n0 + row; gn = gn < N ? gn : N - 1;
        float4 v = *(const float4*)(Bm + (long)gn * ldb_n + (k0 + c4));
        v4h hv; hv[0] = (_Float16)v.x; hv[1] = (_Float16)v.y;
        hv[2] = (_Float16)v.z; hv[3] = (_Float16)v.w;
        *(v4h*)&sB[row][c4] = hv;
      }
    } else if (vecBn && fullK) {
      #pragma unroll
      for (int j = 0; j < 2; ++j) {
        int f = tid * 2 + j;
        int kk = f >> 4, n4 = (f & 15) * 4;   // 4 consecutive n at fixed k
        int gn = n0 + n4; gn = gn <= N - 4 ? gn : N - 4;  // aligned group clamp
        float4 v = *(const float4*)(Bm + (long)(k0 + kk) * ldb_k + gn);
        sB[n4 + 0][kk] = (_Float16)v.x;
        sB[n4 + 1][kk] = (_Float16)v.y;
        sB[n4 + 2][kk] = (_Float16)v.z;
        sB[n4 + 3][kk] = (_Float16)v.w;
      }
    } else {
      #pragma unroll
      for (int r = 0; r < 8; ++r) {
        int i = r * 256 + tid;
        int row = i >> 5, col = i & 31;       // row = n, col = k
        int gn = n0 + row; gn = gn < N ? gn : N - 1;
        int gk = k0 + col;
        int ck = gk < K ? gk : K - 1;
        float bv = Bm[(long)gn * ldb_n + (long)ck * ldb_k];
        sB[row][col] = (_Float16)bv;
        if (gk >= K) sB[row][col] = (_Float16)0.f;
      }
    }
    __syncthreads();
    // prefetch next A K-tile (speculative)
    if (k0 + 32 < K) {
      int pm = m0 + lane; if (pm >= M) pm = M - 1;
      __builtin_prefetch((const void*)&A[(long)pm * lda_m + (long)(k0 + 32) * lda_k], 0, 1);
    }
    // ---- fragments per ISA 7.12.2 16-bit layout: lanes 0-15 hold K 0..7 and
    // 16..23; lanes 16-31 hold K 8..15 and 24..31 (A and transposed-B alike).
    int half = (lane >> 4) << 3;
    int ar = wr * 32 + (lane & 15);
    int br = wc * 16 + (lane & 15);
    v8h a00 = *(const v8h*)&sA[ar][half];
    v8h a01 = *(const v8h*)&sA[ar][16 + half];
    v8h a10 = *(const v8h*)&sA[ar + 16][half];
    v8h a11 = *(const v8h*)&sA[ar + 16][16 + half];
    v8h b0  = *(const v8h*)&sB[br][half];
    v8h b1  = *(const v8h*)&sB[br][16 + half];
    v16h bf  = __builtin_shufflevector(b0, b1, 0,1,2,3,4,5,6,7,8,9,10,11,12,13,14,15);
    v16h af0 = __builtin_shufflevector(a00, a01, 0,1,2,3,4,5,6,7,8,9,10,11,12,13,14,15);
    v16h af1 = __builtin_shufflevector(a10, a11, 0,1,2,3,4,5,6,7,8,9,10,11,12,13,14,15);
    acc0 = __builtin_amdgcn_wmma_f32_16x16x32_f16(
        false, af0, false, bf, (short)0, acc0, false, false);
    acc1 = __builtin_amdgcn_wmma_f32_16x16x32_f16(
        false, af1, false, bf, (short)0, acc1, false, false);
  }

  int colC    = n0 + wc * 16 + (lane & 15);
  int rowBase = m0 + wr * 32 + ((lane >> 4) << 3);
  if (colC < N) {
    float bsv = bias ? bias[colC] : 0.f;
    #pragma unroll
    for (int r = 0; r < 8; ++r) {
      int rowC = rowBase + r;
      if (rowC < M) {
        float v = acc0[r] * alpha + bsv;
        if (act == 1)      v = fmaxf(v, 0.f);
        else if (act == 2) v = 1.f / (1.f + __expf(-v));
        C[(long)rowC * ldc + colC] = v;
      }
      int rowC1 = rowC + 16;
      if (rowC1 < M) {
        float v = acc1[r] * alpha + bsv;
        if (act == 1)      v = fmaxf(v, 0.f);
        else if (act == 2) v = 1.f / (1.f + __expf(-v));
        C[(long)rowC1 * ldc + colC] = v;
      }
    }
  }
}

// ---------------------------------------------------------------------------
// Row softmax (in-place), block per row, 256 threads; explicit row stride.
// ---------------------------------------------------------------------------
__global__ void __launch_bounds__(256) softmax_rows_kernel(float* __restrict__ x,
                                                           int cols, long ldr)
{
  long row = blockIdx.x;
  float* p = x + row * ldr;
  int c = threadIdx.x;
  __shared__ float red[256];
  float mx = -1e30f;
  for (int i = c; i < cols; i += 256) mx = fmaxf(mx, p[i]);
  red[c] = mx; __syncthreads();
  for (int s = 128; s > 0; s >>= 1) { if (c < s) red[c] = fmaxf(red[c], red[c + s]); __syncthreads(); }
  mx = red[0]; __syncthreads();
  float sm = 0.f;
  for (int i = c; i < cols; i += 256) { float e = __expf(p[i] - mx); p[i] = e; sm += e; }
  red[c] = sm; __syncthreads();
  for (int s = 128; s > 0; s >>= 1) { if (c < s) red[c] += red[c + s]; __syncthreads(); }
  float inv = 1.f / red[0];
  for (int i = c; i < cols; i += 256) p[i] *= inv;
}

// ---------------------------------------------------------------------------
// out = layernorm(x + r) * g + b   -- block per token, 256 threads (D=256)
// ---------------------------------------------------------------------------
__global__ void __launch_bounds__(256) add_ln_kernel(
    const float* __restrict__ x, const float* __restrict__ r,
    const float* __restrict__ g, const float* __restrict__ bb, float* __restrict__ out)
{
  long t = blockIdx.x; int c = threadIdx.x;
  float v = x[t * 256 + c] + r[t * 256 + c];
  __shared__ float red[256];
  red[c] = v; __syncthreads();
  for (int s = 128; s > 0; s >>= 1) { if (c < s) red[c] += red[c + s]; __syncthreads(); }
  float mean = red[0] * (1.f / 256.f); __syncthreads();
  float d = v - mean;
  red[c] = d * d; __syncthreads();
  for (int s = 128; s > 0; s >>= 1) { if (c < s) red[c] += red[c + s]; __syncthreads(); }
  float var = red[0] * (1.f / 256.f);
  out[t * 256 + c] = d * rsqrtf(var + 1e-5f) * g[c] + bb[c];
}

__global__ void add2_kernel(const float* a, const float* b, float* o, long n)
{
  long i = blockIdx.x * (long)blockDim.x + threadIdx.x;
  if (i < n) o[i] = a[i] + b[i];
}

__global__ void copy_batched_kernel(float* dst, const float* src,
                                    long dstStride, long srcStride, long n, int nb)
{
  long i = blockIdx.x * (long)blockDim.x + threadIdx.x;
  long total = (long)nb * n;
  if (i >= total) return;
  long z = i / n, r = i - z * n;
  dst[z * dstStride + r] = src[z * srcStride + r];
}

__global__ void build_enh_kernel(const float* qe, const float* qp, const float* te,
                                 float* enh, float* qpos, long total)
{
  long i = blockIdx.x * (long)blockDim.x + threadIdx.x;
  if (i >= total) return;
  int c = (int)(i & 255);
  int t = (int)((i >> 8) % NQ_);
  int ti = t < NSEG_ ? 0 : (t < NSEG_ + NDET_ ? 1 : 2);
  float qpv = qp[t * 256 + c];
  enh[i]  = qe[t * 256 + c] + qpv + te[ti * 256 + c];
  qpos[i] = qpv;
}

__global__ void build_pixq_kernel(const float* sq, const float* sp, float* q, long total)
{
  long i = blockIdx.x * (long)blockDim.x + threadIdx.x;
  if (i >= total) return;
  int c = (int)(i & 255);
  int t = (int)((i >> 8) % 100);
  q[i] = sq[t * 256 + c] + sp[t * 256 + c];
}

__global__ void build_refin_kernel(const float* ref4, const float* vr, float* refin, long total)
{
  long i = blockIdx.x * (long)blockDim.x + threadIdx.x;
  if (i >= total) return;
  int j   = (int)(i & 3);
  int lvl = (int)((i >> 2) & 3);
  long bq = i >> 4;
  int b   = (int)(bq / NQ_);
  refin[i] = ref4[bq * 4 + j] * vr[(b * 4 + lvl) * 2 + (j & 1)];
}

// ---------------------------------------------------------------------------
// MS deformable attention sampling: block per (b,q), thread = (head, dh)
// ---------------------------------------------------------------------------
__global__ void __launch_bounds__(256) ms_deform_sample_kernel(
    const float* __restrict__ value, const float* __restrict__ offs,
    const float* __restrict__ aw, const float* __restrict__ refin,
    float* __restrict__ out)
{
  const int SP_H[4] = {60, 30, 15, 8};
  const int SP_W[4] = {80, 40, 20, 10};
  const int SP_S[4] = {0, 4800, 6000, 6300};
  long bq = blockIdx.x;
  int b = (int)(bq / NQ_);
  int t = threadIdx.x; int h = t >> 5, dc = t & 31;
  const float* rp = refin + bq * 16;
  const float* op = offs + bq * 256;
  const float* ap = aw + bq * 128;
  float acc = 0.f;
  for (int lvl = 0; lvl < 4; ++lvl) {
    int Hh = SP_H[lvl], Ww = SP_W[lvl];
    float rx = rp[lvl * 4 + 0], ry = rp[lvl * 4 + 1];
    float rw = rp[lvl * 4 + 2], rh = rp[lvl * 4 + 3];
    const float* vb = value + ((long)b * LEN_ + SP_S[lvl]) * 256 + h * 32 + dc;
    for (int p = 0; p < 4; ++p) {
      float ox = op[((h * 4 + lvl) * 4 + p) * 2 + 0];
      float oy = op[((h * 4 + lvl) * 4 + p) * 2 + 1];
      float px = (rx + ox * 0.125f * rw) * Ww - 0.5f;
      float py = (ry + oy * 0.125f * rh) * Hh - 0.5f;
      float x0f = floorf(px), y0f = floorf(py);
      int x0 = (int)x0f, y0 = (int)y0f;
      float wx = px - x0f, wy = py - y0f;
      float s = 0.f;
      #pragma unroll
      for (int dy = 0; dy < 2; ++dy)
        #pragma unroll
        for (int dx = 0; dx < 2; ++dx) {
          int yy = y0 + dy, xx = x0 + dx;
          if (yy >= 0 && yy < Hh && xx >= 0 && xx < Ww) {
            float wgt = (dx ? wx : 1.f - wx) * (dy ? wy : 1.f - wy);
            s += wgt * vb[((long)yy * Ww + xx) * 256];
          }
        }
      acc += ap[h * 16 + lvl * 4 + p] * s;
    }
  }
  out[bq * 256 + h * 32 + dc] = acc;
}

__global__ void mean_tokens_kernel(const float* x, float* out, int T)
{
  int i = blockIdx.x * blockDim.x + threadIdx.x;
  if (i >= B_ * 256) return;
  int b = i >> 8, c = i & 255;
  float s = 0.f;
  for (int t = 0; t < T; ++t) s += x[((long)b * T + t) * 256 + c];
  out[i] = s / (float)T;
}

__global__ void point_combine_kernel(const float* pt, const float* s2p, const float* d2p, float* out)
{
  int i = blockIdx.x * blockDim.x + threadIdx.x;
  if (i >= B_ * 3 * 256) return;
  int c = i & 255; int b = i / (3 * 256);
  out[i] = pt[i] + s2p[b * 256 + c] + d2p[b * 256 + c];
}

__global__ void pool_kernel(const float* fused, float* fm, long total)
{
  long i = blockIdx.x * (long)blockDim.x + threadIdx.x;
  if (i >= total) return;
  int o  = (int)(i % 70);
  int dc = (int)((i / 70) % 256);
  int b  = (int)(i / (70 * 256));
  int s = o * 100 / 70;
  int e = ((o + 1) * 100 + 69) / 70;
  float acc = 0.f;
  for (int l = s; l < e; ++l) acc += fused[((long)b * 100 + l) * 256 + dc];
  fm[i] = acc / (float)(e - s);
}

// ConvTranspose2d(stride=2, kernel=4, pad=1); weight layout (Cin, Cout, 4, 4)
__global__ void conv_t_kernel(const float* __restrict__ in, const float* __restrict__ w,
                              const float* __restrict__ bias, float* __restrict__ out,
                              int Cin, int Cout, int H, int W, int relu_f)
{
  int OH = 2 * H, OW = 2 * W;
  long total = (long)B_ * Cout * OH * OW;
  long i = blockIdx.x * (long)blockDim.x + threadIdx.x;
  if (i >= total) return;
  int x  = (int)(i % OW);
  int y  = (int)((i / OW) % OH);
  int co = (int)((i / ((long)OW * OH)) % Cout);
  int b  = (int)(i / ((long)OW * OH * Cout));
  float acc = bias[co];
  for (int ky = 0; ky < 4; ++ky) {
    int ty = y + 1 - ky; if (ty < 0 || (ty & 1)) continue;
    int iy = ty >> 1; if (iy >= H) continue;
    for (int kx = 0; kx < 4; ++kx) {
      int tx = x + 1 - kx; if (tx < 0 || (tx & 1)) continue;
      int ix = tx >> 1; if (ix >= W) continue;
      const float* ip = in + (long)b * Cin * H * W + (long)iy * W + ix;
      const float* wp = w + (long)co * 16 + ky * 4 + kx;
      for (int ci = 0; ci < Cin; ++ci)
        acc += ip[(long)ci * H * W] * wp[(long)ci * Cout * 16];
    }
  }
  if (relu_f) acc = fmaxf(acc, 0.f);
  out[i] = acc;
}

__global__ void resize_kernel(const float* in, float* out, int BC, int IH, int IW, int OH, int OW)
{
  long total = (long)BC * OH * OW;
  long i = blockIdx.x * (long)blockDim.x + threadIdx.x;
  if (i >= total) return;
  int ox = (int)(i % OW);
  int oy = (int)((i / OW) % OH);
  int bc = (int)(i / ((long)OW * OH));
  float sy = (oy + 0.5f) * (float)IH / OH - 0.5f;
  float sx = (ox + 0.5f) * (float)IW / OW - 0.5f;
  sy = fminf(fmaxf(sy, 0.f), (float)IH - 1.f);
  sx = fminf(fmaxf(sx, 0.f), (float)IW - 1.f);
  int y0 = (int)sy, x0 = (int)sx;
  int y1 = min(y0 + 1, IH - 1), x1 = min(x0 + 1, IW - 1);
  float fy = sy - y0, fx = sx - x0;
  const float* p = in + (long)bc * IH * IW;
  out[i] = p[(long)y0 * IW + x0] * (1 - fx) * (1 - fy)
         + p[(long)y0 * IW + x1] * fx * (1 - fy)
         + p[(long)y1 * IW + x0] * (1 - fx) * fy
         + p[(long)y1 * IW + x1] * fx * fy;
}

// ---------------------------------------------------------------------------
extern "C" void kernel_launch(void* const* d_in, const int* in_sizes, int n_in,
                              void* d_out, int out_size, void* d_ws, size_t ws_size,
                              hipStream_t stream)
{
  (void)in_sizes; (void)n_in; (void)out_size; (void)ws_size;
  const float* Vfeat = (const float*)d_in[0];
  const float* vr    = (const float*)d_in[3];

  // ---- unpack params (jax tree flatten: sorted dict keys, lists in order) ----
  int idx = 4;
  auto nxt = [&]() { return (const float*)d_in[idx++]; };
  struct LinP { const float* w; const float* b; };
  struct LNP  { const float* g; const float* b; };
  struct MHAP { LinP k, o, q, v; };
  auto getLin = [&]() { LinP l; l.b = nxt(); l.w = nxt(); return l; };
  auto getLN  = [&]() { LNP n; n.b = nxt(); n.g = nxt(); return n; };
  auto getMHA = [&]() { MHAP m; m.k = getLin(); m.o = getLin(); m.q = getLin(); m.v = getLin(); return m; };
  struct CAP  { LinP aw, outp, soff, vproj; };
  struct DLay { CAP ca; LinP f1, f2; LNP n1, n2, n3; MHAP sa; };
  auto getDL = [&]() {
    DLay d;
    d.ca.aw = getLin(); d.ca.outp = getLin(); d.ca.soff = getLin(); d.ca.vproj = getLin();
    d.f1 = getLin(); d.f2 = getLin(); d.n1 = getLN(); d.n2 = getLN(); d.n3 = getLN();
    d.sa = getMHA(); return d;
  };
  struct PLay { MHAP ca; LinP f1, f2; LNP n1, n2, n3; MHAP sa; };
  auto getPL = [&]() {
    PLay p; p.ca = getMHA(); p.f1 = getLin(); p.f2 = getLin();
    p.n1 = getLN(); p.n2 = getLN(); p.n3 = getLN(); p.sa = getMHA(); return p;
  };

  MHAP ct_attn = getMHA();
  LNP  ct_norm = getLN();
  DLay dls[6]; for (int i = 0; i < 6; ++i) dls[i] = getDL();
  LinP det_bbox1 = getLin(), det_bbox2 = getLin();
  LinP det_class1 = getLin(), det_class2 = getLin();
  LinP det_to_point = getLin();
  LinP fuse1 = getLin(), fuse2 = getLin();
  PLay pls[2]; pls[0] = getPL(); pls[1] = getPL();
  const float* seg_pos = nxt(); const float* seg_queries = nxt();
  LinP up[5]; for (int i = 0; i < 5; ++i) up[i] = getLin();
  LinP point1 = getLin(), point2 = getLin();
  const float* query_embed = nxt(); const float* query_pos = nxt();
  LinP refpts = getLin();
  LinP seg_head1 = getLin(), seg_head2 = getLin(), seg_to_point = getLin();
  const float* task_embed = nxt();

  // ---- output regions (return-order concat) ----
  float* out = (float*)d_out;
  float* o_segq   = out;
  float* o_seglog = o_segq + 8L * 256 * 256;
  float* o_det    = o_seglog + 8L * 10 * 640 * 480;
  float* o_cls    = o_det + 8L * 256 * 256;
  float* o_bbox   = o_cls + 8L * 256 * 11;
  float* o_point  = o_bbox + 8L * 256 * 4;
  float* o_pc     = o_point + 8L * 3 * 256;
  float* o_uni    = o_pc + 8L * 3 * 7;

  // ---- workspace bump allocator ----
  char* wsBase = (char*)d_ws; size_t wsOff = 0;
  auto alloc = [&](long elems) {
    float* p = (float*)(wsBase + wsOff);
    wsOff += ((size_t)elems * 4 + 255) & ~(size_t)255;
    return p;
  };

  // ---- launch helpers ----
  auto gemm = [&](const float* A, const float* Bm, const float* bias, float* C,
                  int M, int N, int K,
                  long lda_m, long lda_k, long ldb_n, long ldb_k, long ldc,
                  int nb, int nh, long sAb, long sAh, long sBb, long sBh,
                  long sCb, long sCh, float alpha, int act) {
    dim3 g((N + 63) / 64, (M + 63) / 64, nb);
    wmma_gemm_kernel<<<g, dim3(256), 0, stream>>>(A, Bm, bias, C, M, N, K,
        lda_m, lda_k, ldb_n, ldb_k, ldc, nh, sAb, sAh, sBb, sBh, sCb, sCh, alpha, act);
  };
  auto lin = [&](const float* X, int M, const LinP& L, int N, int K, float* Y, int act) {
    gemm(X, L.w, L.b, Y, M, N, K, K, 1, K, 1, N, 1, 1, 0, 0, 0, 0, 0, 0, 1.f, act);
  };
  auto addln = [&](const float* x, const float* r, const LNP& n, float* o2, int ntok) {
    add_ln_kernel<<<ntok, 256, 0, stream>>>(x, r, n.g, n.b, o2);
  };
  auto add2 = [&](const float* a, const float* b, float* o2, long n) {
    add2_kernel<<<(int)((n + 255) / 256), 256, 0, stream>>>(a, b, o2, n);
  };
  auto copyb = [&](float* dst, const float* src, long ds, long ss, long n, int nb) {
    copy_batched_kernel<<<(int)(((long)nb * n + 255) / 256), 256, 0, stream>>>(dst, src, ds, ss, n, nb);
  };
  // scores rows padded to multiple of 4 so the P@V GEMM's A qualifies for the
  // aligned float4 staging path.
  auto mha = [&](const float* Qin, const float* Kin, const float* Vin, const MHAP& P,
                 int Lq, int Lk, float* Qp, float* Kp, float* Vp, float* sc,
                 float* ctx, float* outp) {
    long ldS = (Lk + 3) & ~3L;
    lin(Qin, B_ * Lq, P.q, 256, 256, Qp, 0);
    lin(Kin, B_ * Lk, P.k, 256, 256, Kp, 0);
    lin(Vin, B_ * Lk, P.v, 256, 256, Vp, 0);
    gemm(Qp, Kp, nullptr, sc, Lq, Lk, 32, 256, 1, 256, 1, ldS,
         B_ * NH_, NH_, (long)Lq * 256, 32, (long)Lk * 256, 32,
         (long)NH_ * Lq * ldS, (long)Lq * ldS, 0.17677669529663687f, 0);
    softmax_rows_kernel<<<B_ * NH_ * Lq, 256, 0, stream>>>(sc, Lk, ldS);
    gemm(sc, Vp, nullptr, ctx, Lq, 32, Lk, ldS, 1, 1, 256, 256,
         B_ * NH_, NH_, (long)NH_ * Lq * ldS, (long)Lq * ldS,
         (long)Lk * 256, 32, (long)Lq * 256, 32, 1.f, 0);
    lin(ctx, B_ * Lq, P.o, 256, 256, outp, 0);
  };

  // ---- persistent buffers ----
  float* enh   = alloc(8L * NQ_ * 256);
  float* qpos  = alloc(8L * NQ_ * 256);
  float* tgt   = alloc(8L * NQ_ * 256);
  float* ref4  = alloc(8L * NQ_ * 4);
  float* refin = alloc(8L * NQ_ * 16);
  size_t ckpt = wsOff;

  // ---- decoder-phase scratch ----
  const long ldSd = (NQ_ + 3) & ~3L;  // 516
  float* tq     = alloc(8L * NQ_ * 256);
  float* Qp     = alloc(8L * NQ_ * 256);
  float* Kp     = alloc(8L * NQ_ * 256);
  float* Vp     = alloc(8L * NQ_ * 256);
  float* ctx    = alloc(8L * NQ_ * 256);
  float* proj   = alloc(8L * NQ_ * 256);
  float* scores = alloc(8L * NH_ * NQ_ * ldSd);
  float* value  = alloc(8L * LEN_ * 256);
  float* ffnh   = alloc(8L * NQ_ * 1024);
  float* offsb  = alloc(8L * NQ_ * 256);
  float* awb    = alloc(8L * NQ_ * 128);
  float* segb   = alloc(8L * 256 * 256);
  float* detb   = alloc(8L * 256 * 256);
  float* ptb    = alloc(8L * 3 * 256);
  float* segmean = alloc(8L * 256);
  float* detmean = alloc(8L * 256);
  float* s2pb   = alloc(8L * 256);
  float* d2pb   = alloc(8L * 256);
  float* pointb = alloc(8L * 3 * 256);
  float* ph     = alloc(8L * 3 * 256);

  // ---- build enhanced queries, reference points ----
  long totE = 8L * NQ_ * 256;
  build_enh_kernel<<<(int)((totE + 255) / 256), 256, 0, stream>>>(
      query_embed, query_pos, task_embed, enh, qpos, totE);
  lin(enh, B_ * NQ_, refpts, 4, 256, ref4, 2);  // sigmoid
  long totR = 8L * NQ_ * 16;
  build_refin_kernel<<<(int)((totR + 255) / 256), 256, 0, stream>>>(ref4, vr, refin, totR);
  copyb(tgt, enh, 0, 0, totE, 1);

  // ---- 6 deformable decoder layers ----
  for (int l = 0; l < 6; ++l) {
    const DLay& L = dls[l];
    add2(tgt, qpos, tq, totE);
    mha(tq, tq, tgt, L.sa, NQ_, NQ_, Qp, Kp, Vp, scores, ctx, proj);
    addln(tgt, proj, L.n2, tgt, B_ * NQ_);
    add2(tgt, qpos, tq, totE);
    lin(Vfeat, B_ * LEN_, L.ca.vproj, 256, 256, value, 0);
    lin(tq, B_ * NQ_, L.ca.soff, 256, 256, offsb, 0);
    lin(tq, B_ * NQ_, L.ca.aw, 128, 256, awb, 0);
    softmax_rows_kernel<<<B_ * NQ_ * NH_, 256, 0, stream>>>(awb, 16, 16);
    ms_deform_sample_kernel<<<B_ * NQ_, 256, 0, stream>>>(value, offsb, awb, refin, ctx);
    lin(ctx, B_ * NQ_, L.ca.outp, 256, 256, proj, 0);
    addln(tgt, proj, L.n1, tgt, B_ * NQ_);
    lin(tgt, B_ * NQ_, L.f1, 1024, 256, ffnh, 1);
    lin(ffnh, B_ * NQ_, L.f2, 256, 1024, proj, 0);
    addln(tgt, proj, L.n3, tgt, B_ * NQ_);
  }

  // ---- unified / split ----
  copyb(o_uni, tgt, 0, 0, totE, 1);
  copyb(segb, tgt, 256L * 256, (long)NQ_ * 256, 256L * 256, B_);
  copyb(detb, tgt + 256L * 256, 256L * 256, (long)NQ_ * 256, 256L * 256, B_);
  copyb(ptb, tgt + 512L * 256, 3L * 256, (long)NQ_ * 256, 3L * 256, B_);

  // ---- cross-task attention ----
  mha(segb, detb, detb, ct_attn, 256, 256, Qp, Kp, Vp, scores, ctx, proj);
  addln(segb, proj, ct_norm, segb, B_ * 256);
  mha(detb, segb, segb, ct_attn, 256, 256, Qp, Kp, Vp, scores, ctx, proj);
  addln(detb, proj, ct_norm, detb, B_ * 256);
  copyb(o_det, detb, 0, 0, 8L * 256 * 256, 1);

  // ---- point path ----
  mean_tokens_kernel<<<(B_ * 256 + 255) / 256, 256, 0, stream>>>(segb, segmean, 256);
  mean_tokens_kernel<<<(B_ * 256 + 255) / 256, 256, 0, stream>>>(detb, detmean, 256);
  lin(segmean, B_, seg_to_point, 256, 256, s2pb, 0);
  lin(detmean, B_, det_to_point, 256, 256, d2pb, 0);
  point_combine_kernel<<<(B_ * 3 * 256 + 255) / 256, 256, 0, stream>>>(ptb, s2pb, d2pb, pointb);
  copyb(o_point, pointb, 0, 0, 8L * 3 * 256, 1);
  lin(pointb, B_ * 3, point1, 256, 256, ph, 1);
  lin(ph, B_ * 3, point2, 7, 256, o_pc, 0);

  // ---- detection heads ----
  lin(detb, B_ * 256, det_class1, 256, 256, ffnh, 1);
  lin(ffnh, B_ * 256, det_class2, 11, 256, o_cls, 0);
  lin(detb, B_ * 256, det_bbox1, 256, 256, ffnh, 1);
  lin(ffnh, B_ * 256, det_bbox2, 4, 256, o_bbox, 2);  // sigmoid

  // ---- segmentation head ----
  lin(segb, B_ * 256, seg_head1, 256, 256, ffnh, 1);
  lin(ffnh, B_ * 256, seg_head2, 256, 256, o_segq, 0);

  // ---- pixel decoder (reuse decoder scratch region) ----
  wsOff = ckpt;
  const int LM = NSEG_ + LEN_;  // 6636 (multiple of 4)
  float* pq   = alloc(8L * 100 * 256);
  float* mem  = alloc(8L * LM * 256);
  float* Qp2  = alloc(8L * 100 * 256);
  float* Kp2  = alloc(8L * LM * 256);
  float* Vp2  = alloc(8L * LM * 256);
  float* sc2  = alloc(8L * NH_ * 100 * LM);
  float* ctx2 = alloc(8L * 100 * 256);
  float* pr2  = alloc(8L * 100 * 256);
  float* fh2  = alloc(8L * 100 * 512);
  float* fused = alloc(8L * 100 * 256);
  float* fm1  = alloc(8L * 10 * 224 * 320);
  float* fm2  = alloc(8L * 16 * 112 * 160);

  long totP = 8L * 100 * 256;
  build_pixq_kernel<<<(int)((totP + 255) / 256), 256, 0, stream>>>(seg_queries, seg_pos, pq, totP);
  copyb(mem, o_segq, (long)LM * 256, 256L * 256, 256L * 256, B_);
  copyb(mem + 256L * 256, Vfeat, (long)LM * 256, (long)LEN_ * 256, (long)LEN_ * 256, B_);

  for (int l = 0; l < 2; ++l) {
    const PLay& P = pls[l];
    mha(pq, pq, pq, P.sa, 100, 100, Qp2, Kp2, Vp2, sc2, ctx2, pr2);
    addln(pq, pr2, P.n1, pq, B_ * 100);
    mha(pq, mem, mem, P.ca, 100, LM, Qp2, Kp2, Vp2, sc2, ctx2, pr2);
    addln(pq, pr2, P.n2, pq, B_ * 100);
    lin(pq, B_ * 100, P.f1, 512, 256, fh2, 1);
    lin(fh2, B_ * 100, P.f2, 256, 512, pr2, 0);
    addln(pq, pr2, P.n3, pq, B_ * 100);
  }

  lin(pq, B_ * 100, fuse1, 256, 256, ctx2, 1);
  lin(ctx2, B_ * 100, fuse2, 256, 256, fused, 0);
  long totF = 8L * 256 * 70;
  pool_kernel<<<(int)((totF + 255) / 256), 256, 0, stream>>>(fused, fm1, totF);

  int chans[6] = {256, 128, 64, 32, 16, 10};
  int H = 7, W = 10;
  float* fa = fm1; float* fb = fm2;
  for (int s = 0; s < 5; ++s) {
    long total = 8L * chans[s + 1] * (2 * H) * (2 * W);
    conv_t_kernel<<<(int)((total + 255) / 256), 256, 0, stream>>>(
        fa, up[s].w, up[s].b, fb, chans[s], chans[s + 1], H, W, (s < 4) ? 1 : 0);
    float* t = fa; fa = fb; fb = t; H *= 2; W *= 2;
  }
  long totZ = 8L * 10 * 640 * 480;
  resize_kernel<<<(int)((totZ + 255) / 256), 256, 0, stream>>>(fa, o_seglog, 8 * 10, 224, 320, 640, 480);
}
